// Part_Conv_Att_17008070492973
// MI455X (gfx1250) — matrix operand order repaired and checked by reading, not compiled, not run on hardware
//
#include <hip/hip_runtime.h>
#include <hip/hip_bf16.h>

// ---------------------------------------------------------------------------
// Part_Conv_Att for MI455X (gfx1250, wave32, WMMA)
//
// Stage 1: part 1x1-conv GEMM (K=256 -> 64 ch) via v_wmma_f32_16x16x32_bf16,
//          fused bias+BN+ReLU+mean-pool epilogue -> pooled[N][64] (atomics).
//          A fragments packed f32->bf16 with v_perm_b32 (1 perm / 2 elems).
//          Out-of-range tile rows are clamped (not predicated) and masked in
//          the epilogue, keeping the inner loop branch/EXEC-free.
// Stage 2: pooled -> fc1 -> BN -> ReLU -> fc2 -> softmax(P=5) -> att[N][256][5]
// Stage 3: out = relu(x * (att[joint]*bn_scale + 1) + bn_shift)  (float4 I/O)
// ---------------------------------------------------------------------------

typedef __attribute__((ext_vector_type(8)))  float          v8f;
typedef __attribute__((ext_vector_type(16))) __bf16         v16bf;
typedef __attribute__((ext_vector_type(16))) unsigned short v16us;
typedef __attribute__((ext_vector_type(8)))  unsigned int   v8u;

#define NN    64
#define CC    256
#define CQ    64
#define TT    300
#define VV    17
#define TVS   5100      // T*V
#define NPART 5
#define EPS   1e-5f

__device__ __constant__ int kJp[NPART]            = {5, 3, 3, 3, 3};
__device__ __constant__ int kJoints[NPART][5]     = {
    {0, 1, 2, 3, 4}, {5, 7, 9, 0, 0}, {6, 8, 10, 0, 0},
    {11, 13, 15, 0, 0}, {12, 14, 16, 0, 0}};
__device__ __constant__ int kJointOf[VV] = {0,0,0,0,0,1,2,1,2,1,2,3,4,3,4,3,4};

// float -> bf16 (round to nearest even), scalar (used for weight staging only)
__device__ __forceinline__ unsigned short f2bf(float f) {
    unsigned int u = __float_as_uint(f);
    u += 0x7FFFu + ((u >> 16) & 1u);
    return (unsigned short)(u >> 16);
}

// Pack two floats into {bf16(f1),bf16(f0)} with one v_perm_b32.
// Round-to-nearest via +0x8000 bias before truncating to the high half.
__device__ __forceinline__ unsigned int pack_bf16x2(float f0, float f1) {
    unsigned int u0 = __float_as_uint(f0) + 0x8000u;
    unsigned int u1 = __float_as_uint(f1) + 0x8000u;
    // sel bytes (LSB..MSB): 2,3 from src_b (u0) ; 6,7 from src_a (u1)
    return __builtin_amdgcn_perm(u1, u0, 0x07060302u);
}

// ---------------------------------------------------------------------------
__global__ void zero_ws_kernel(float* __restrict__ p, int n) {
    int i = blockIdx.x * 256 + threadIdx.x;
    if (i < n) p[i] = 0.f;
}

// ---------------------------------------------------------------------------
// Stage 1: one wave = one 16-row x 64-col output tile of part p.
// Rows enumerate (t, joint-in-part) for one sample n.
// ---------------------------------------------------------------------------
__global__ __launch_bounds__(256) void stage1_partconv(
    const float* __restrict__ x, const float* __restrict__ part_w,
    const float* __restrict__ part_b, const float* __restrict__ pbn_g,
    const float* __restrict__ pbn_b, const float* __restrict__ pbn_m,
    const float* __restrict__ pbn_v, float* __restrict__ pooled)
{
    __shared__ unsigned short Wlds[CQ * CC];   // 32 KB bf16 weights for part p

    const int p   = blockIdx.x;
    const int tid = threadIdx.x;

    // Stage weights (64x256 f32 -> bf16) into LDS once per block.
    const float* wp = part_w + (size_t)p * CQ * CC;
    for (int i = tid; i < CQ * CC; i += 256) Wlds[i] = f2bf(wp[i]);
    __syncthreads();

    const int Jp          = kJp[p];
    const int Mp          = TT * Jp;              // rows per sample for this part
    const int tiles_per_n = (Mp + 15) >> 4;
    const int wave  = tid >> 5;
    const int lane  = tid & 31;
    const int gtile = blockIdx.y * 8 + wave;
    if (gtile >= NN * tiles_per_n) return;        // after barrier: safe

    const int n     = gtile / tiles_per_n;
    const int mbase = (gtile % tiles_per_n) << 4;

    // ---- A-fragment geometry (16-bit A 16x32 layout) ----
    // Clamp out-of-range rows to a valid address; their (finite) garbage only
    // lands in D rows that the epilogue masks out of the pooled sum.
    const int lrow = lane & 15;                   // A row = D column index
    const int mA   = mbase + lrow;
    const int mC   = (mA < Mp) ? mA : (Mp - 1);
    int t, v;
    if (Jp == 5) { t = mC / 5; v = kJoints[p][mC % 5]; }
    else         { t = mC / 3; v = kJoints[p][mC % 3]; }
    const float* xr  = x + (size_t)n * CC * TVS + (size_t)t * VV + v;
    const int khA = (lane >> 4) << 4;             // lanes16-31 hold K+16 block
    const int khB = (lane >> 4) << 4;             // lanes16-31 hold K+16 block

    v8f acc0 = (v8f)0.f, acc1 = (v8f)0.f, acc2 = (v8f)0.f, acc3 = (v8f)0.f;

#pragma unroll 2
    for (int ko = 0; ko < CC; ko += 32) {
        // A fragment: 16 strided dword loads along channel dim,
        // packed to bf16 pairs with v_perm_b32 (branch-free).
        v8u au;
#pragma unroll
        for (int i = 0; i < 4; i++) {
            const int k0 = ko + khA + 2 * i;        // fragment dwords 0..3
            au[i] = pack_bf16x2(xr[(size_t)(k0    ) * TVS],
                                xr[(size_t)(k0 + 1) * TVS]);
            const int k1 = ko + 8 + khA + 2 * i;    // fragment dwords 4..7
            au[4 + i] = pack_bf16x2(xr[(size_t)(k1    ) * TVS],
                                    xr[(size_t)(k1 + 1) * TVS]);
        }
        v16bf a = __builtin_bit_cast(v16bf, au);

        // B fragments: 16 contiguous bf16 per lane from LDS per q-tile.
#pragma unroll
        for (int qt = 0; qt < 4; qt++) {
            const v16us bu = *(const v16us*)(&Wlds[(size_t)(qt * 16 + lrow) * CC + ko + khB]);
            v16bf b = __builtin_bit_cast(v16bf, bu);
            switch (qt) {
            case 0: acc0 = __builtin_amdgcn_wmma_f32_16x16x32_bf16(false, a, false, b, (short)0, acc0, false, false); break;
            case 1: acc1 = __builtin_amdgcn_wmma_f32_16x16x32_bf16(false, a, false, b, (short)0, acc1, false, false); break;
            case 2: acc2 = __builtin_amdgcn_wmma_f32_16x16x32_bf16(false, a, false, b, (short)0, acc2, false, false); break;
            case 3: acc3 = __builtin_amdgcn_wmma_f32_16x16x32_bf16(false, a, false, b, (short)0, acc3, false, false); break;
            }
        }
    }

    // ---- Epilogue: bias + BN + ReLU, masked row-sum, atomic pool ----
    const int   rowoff = (lane >> 4) << 3;        // D rows: r + 8*(lane>>4)
    const float invMp  = 1.0f / (float)Mp;        // mean over (T, V_part)

#pragma unroll
    for (int qt = 0; qt < 4; qt++) {
        v8f acc = (qt == 0) ? acc0 : (qt == 1) ? acc1 : (qt == 2) ? acc2 : acc3;
        const int q  = qt * 16 + lrow;
        const int pq = p * CQ + q;
        const float sc  = pbn_g[pq] * rsqrtf(pbn_v[pq] + EPS);
        const float off = part_b[pq] * sc + (pbn_b[pq] - pbn_m[pq] * sc);
        float s = 0.f;
#pragma unroll
        for (int r = 0; r < 8; r++) {
            const int mD = mbase + rowoff + r;
            float y = fmaxf(acc[r] * sc + off, 0.f);
            if (mD < Mp) s += y;
        }
        atomicAdd(&pooled[n * CQ + q], s * invMp);
    }
}

// ---------------------------------------------------------------------------
// Stage 2: per-n FC chain + softmax over P. One block (256 thr) per n.
// ---------------------------------------------------------------------------
__global__ __launch_bounds__(256) void stage2_fc(
    const float* __restrict__ pooled,
    const float* __restrict__ fc1_w, const float* __restrict__ fc1_b,
    const float* __restrict__ fbn_g, const float* __restrict__ fbn_b,
    const float* __restrict__ fbn_m, const float* __restrict__ fbn_v,
    const float* __restrict__ fc2_w, const float* __restrict__ fc2_b,
    float* __restrict__ att)
{
    __shared__ float psh[CQ];
    __shared__ float hsh[CQ];
    const int n = blockIdx.x, tid = threadIdx.x;

    if (tid < CQ) psh[tid] = pooled[n * CQ + tid];
    __syncthreads();

    if (tid < CQ) {
        float s = fc1_b[tid];
        const float* wr = fc1_w + (size_t)tid * CQ;
#pragma unroll 8
        for (int c = 0; c < CQ; c++) s += psh[c] * wr[c];
        const float sc = fbn_g[tid] * rsqrtf(fbn_v[tid] + EPS);
        s = s * sc + (fbn_b[tid] - fbn_m[tid] * sc);
        hsh[tid] = fmaxf(s, 0.f);
    }
    __syncthreads();

    // tid == output channel c (0..255); compute z[c][0..4], softmax over P.
    float z[NPART];
#pragma unroll
    for (int pp = 0; pp < NPART; pp++) {
        const int i = tid * NPART + pp;
        float s = fc2_b[i];
        const float* wr = fc2_w + (size_t)i * CQ;
#pragma unroll 8
        for (int c = 0; c < CQ; c++) s += hsh[c] * wr[c];
        z[pp] = s;
    }
    float mx = z[0];
#pragma unroll
    for (int pp = 1; pp < NPART; pp++) mx = fmaxf(mx, z[pp]);
    float den = 0.f;
#pragma unroll
    for (int pp = 0; pp < NPART; pp++) { z[pp] = __expf(z[pp] - mx); den += z[pp]; }
    const float inv = 1.0f / den;
#pragma unroll
    for (int pp = 0; pp < NPART; pp++)
        att[(size_t)n * (CC * NPART) + tid * NPART + pp] = z[pp] * inv;
}

// ---------------------------------------------------------------------------
// Stage 3: out = relu(x * (att[joint]*bn_scale + 1) + bn_shift).
// One block per (n,c) slice of 5100 contiguous floats; float4 I/O.
// ---------------------------------------------------------------------------
__global__ __launch_bounds__(256) void stage3_apply(
    const float* __restrict__ x, const float* __restrict__ att,
    const float* __restrict__ bn_g, const float* __restrict__ bn_b,
    const float* __restrict__ bn_m, const float* __restrict__ bn_v,
    float* __restrict__ out)
{
    __shared__ float coef[VV];
    __shared__ float shsh;
    const int b = blockIdx.x;
    const int n = b >> 8;
    const int c = b & 255;
    const int tid = threadIdx.x;

    if (tid < VV) {
        const float sc = bn_g[c] * rsqrtf(bn_v[c] + EPS);
        const float a  = att[(size_t)n * (CC * NPART) + c * NPART + kJointOf[tid]];
        coef[tid] = a * sc + 1.0f;
        if (tid == 0) shsh = bn_b[c] - bn_m[c] * sc;
    }
    __syncthreads();

    const float shift = shsh;
    const float4* xp = (const float4*)(x + (size_t)b * TVS);
    float4*       op = (float4*)(out + (size_t)b * TVS);
    for (int e4 = tid; e4 < TVS / 4; e4 += 256) {
        const float4 xv = xp[e4];
        const int e = e4 * 4;
        float4 o;
        o.x = fmaxf(xv.x * coef[(e    ) % VV] + shift, 0.f);
        o.y = fmaxf(xv.y * coef[(e + 1) % VV] + shift, 0.f);
        o.z = fmaxf(xv.z * coef[(e + 2) % VV] + shift, 0.f);
        o.w = fmaxf(xv.w * coef[(e + 3) % VV] + shift, 0.f);
        op[e4] = o;
    }
}

// ---------------------------------------------------------------------------
extern "C" void kernel_launch(void* const* d_in, const int* in_sizes, int n_in,
                              void* d_out, int out_size, void* d_ws, size_t ws_size,
                              hipStream_t stream) {
    const float* x      = (const float*)d_in[0];
    const float* part_w = (const float*)d_in[1];
    const float* part_b = (const float*)d_in[2];
    const float* pbn_g  = (const float*)d_in[3];
    const float* pbn_b  = (const float*)d_in[4];
    const float* pbn_m  = (const float*)d_in[5];
    const float* pbn_v  = (const float*)d_in[6];
    const float* fc1_w  = (const float*)d_in[7];
    const float* fc1_b  = (const float*)d_in[8];
    const float* fbn_g  = (const float*)d_in[9];
    const float* fbn_b  = (const float*)d_in[10];
    const float* fbn_m  = (const float*)d_in[11];
    const float* fbn_v  = (const float*)d_in[12];
    const float* fc2_w  = (const float*)d_in[13];
    const float* fc2_b  = (const float*)d_in[14];
    const float* bn_g   = (const float*)d_in[15];
    const float* bn_b   = (const float*)d_in[16];
    const float* bn_m   = (const float*)d_in[17];
    const float* bn_v   = (const float*)d_in[18];

    float* pooled = (float*)d_ws;                 // N*CQ = 4096 floats
    float* att    = pooled + NN * CQ;             // N*C*P = 81920 floats
    float* outp   = (float*)d_out;

    zero_ws_kernel<<<(NN * CQ + 255) / 256, 256, 0, stream>>>(pooled, NN * CQ);

    // part 0: 94 tiles/n -> 6016 waves -> 752 blocks of 8 waves (max over parts)
    dim3 g1(NPART, 752);
    stage1_partconv<<<g1, 256, 0, stream>>>(x, part_w, part_b, pbn_g, pbn_b,
                                            pbn_m, pbn_v, pooled);

    stage2_fc<<<NN, 256, 0, stream>>>(pooled, fc1_w, fc1_b, fbn_g, fbn_b,
                                      fbn_m, fbn_v, fc2_w, fc2_b, att);

    stage3_apply<<<NN * CC, 256, 0, stream>>>(x, att, bn_g, bn_b, bn_m, bn_v,
                                              outp);
}
